// NCA_7576322310217
// MI455X (gfx1250) — matrix operand order repaired and checked by reading, compiled, not run on hardware
//
#include <hip/hip_runtime.h>
#include <hip/hip_bf16.h>

typedef __attribute__((ext_vector_type(16))) _Float16 v16h;
typedef __attribute__((ext_vector_type(8)))  _Float16 v8h;
typedef __attribute__((ext_vector_type(8)))  float    v8f;

#define HWPIX   784   // 28*28
#define C_CH    16
#define F_CH    128
#define D_CH    128
#define KPAD    160   // 9*16 = 144 padded to 5*32 (k=144 carries the bias column)
#define TSTEPS  50
#define NTILES  49    // 784 / 16

// ---- LDS layout (bytes) ----
// grid f32 : 784*16*4  = 50176   @ 0
// delta    : 784*16*4  = 50176   @ 50176   (also input staging at init)
// grid f16 : 784*16*2  = 25088   @ 100352  (shadow copy for conv A-frags)
// wT conv  : 128*160*2 = 40960   @ 125440  (B-operand layout [f][k], k=144 = bias)
// w1t      : 128*128*2 = 32768   @ 166400  ([d][f])
// w2t      : 16*128*2  = 4096    @ 199168  ([c][d])
// b1       : 128*4     = 512     @ 203264
// b2       : 16*4      = 64      @ 203776
// staging  : 8*16*128*2= 32768   @ 203840  (per-wave h1/h2 tile)
#define SMEM_BYTES 236608

__global__ __launch_bounds__(256) void nca_fused_kernel(
    const float* __restrict__ in,      // (256,28,28)
    const float* __restrict__ conv_w,  // (3,3,16,128) HWIO
    const float* __restrict__ conv_b,  // (128)
    const float* __restrict__ w1,      // (128,128)
    const float* __restrict__ b1,      // (128)
    const float* __restrict__ w2,      // (128,16)
    const float* __restrict__ b2,      // (16)
    float* __restrict__ out)           // (256,28,28,10)
{
    extern __shared__ char smem[];
    float*    g    = (float*)(smem);
    float*    dlt  = (float*)(smem + 50176);
    _Float16* gh   = (_Float16*)(smem + 100352);
    _Float16* wT   = (_Float16*)(smem + 125440);
    _Float16* w1t  = (_Float16*)(smem + 166400);
    _Float16* w2t  = (_Float16*)(smem + 199168);
    float*    b1s  = (float*)(smem + 203264);
    float*    b2s  = (float*)(smem + 203776);
    _Float16* stg  = (_Float16*)(smem + 203840);

    const int b    = blockIdx.x;
    const int tid  = threadIdx.x;
    const int lane = tid & 31;
    const int wv   = tid >> 5;
    const int hs   = lane >> 4;   // half-wave select
    const int lr   = lane & 15;

    // ---- async-stage the per-image input row into LDS (delta buffer is free) ----
    for (int i = tid; i < HWPIX; i += 256) {
        unsigned long long ga = (unsigned long long)(const void*)(in + b * HWPIX + i);
        unsigned int       la = (unsigned int)(unsigned long long)(const void*)(dlt + i);
        asm volatile("global_load_async_to_lds_b32 %0, %1, off"
                     :: "v"(la), "v"(ga) : "memory");
    }
    asm volatile("s_wait_asynccnt 0x0" ::: "memory");

    // ---- one-time weight transposes into LDS (f16, B-fragment friendly) ----
    for (int i = tid; i < F_CH * KPAD; i += 256) {
        int f = i / KPAD, k = i - f * KPAD;
        _Float16 v = (_Float16)0.f;
        if (k < 144) {
            int t = k >> 4, c = k & 15;            // tap, channel
            v = (_Float16)conv_w[(t * 16 + c) * F_CH + f];
        } else if (k == 144) {
            v = (_Float16)conv_b[f];               // bias folded into K
        }
        wT[f * KPAD + k] = v;
    }
    for (int i = tid; i < D_CH * F_CH; i += 256) {
        int d = i >> 7, f = i & 127;
        w1t[d * F_CH + f] = (_Float16)w1[f * D_CH + d];
    }
    for (int i = tid; i < C_CH * D_CH; i += 256) {
        int c = i >> 7, d = i & 127;
        w2t[c * D_CH + d] = (_Float16)w2[d * C_CH + c];
    }
    for (int i = tid; i < F_CH; i += 256) b1s[i] = b1[i];
    if (tid < C_CH) b2s[tid] = b2[tid];
    __syncthreads();   // async-staged input visible to all waves

    // ---- grid init: ch0 = thresholded input, rest 0 (f32 + f16 shadow) ----
    for (int i = tid; i < HWPIX * C_CH; i += 256) {
        int p = i >> 4, c = i & 15;
        float v = 0.f;
        if (c == 0) v = (dlt[p] > 0.1f) ? 1.f : 0.f;
        g[i]  = v;
        gh[i] = (_Float16)v;
    }
    __syncthreads();

    _Float16* myStg = stg + wv * (16 * F_CH);

    for (int step = 0; step < TSTEPS; ++step) {
        for (int tile = wv; tile < NTILES; tile += 8) {
            const int p = tile * 16 + lr;
            const int y = p / 28, x = p - y * 28;

            // ---- build conv A-fragments from f16 shadow grid (5 K-chunks) ----
            v16h aF[5];
            #pragma unroll
            for (int kc = 0; kc < 5; ++kc) {
                const int k0 = kc * 32;
                v16h a;
                #pragma unroll
                for (int half = 0; half < 2; ++half) {
                    const int tap = (k0 >> 4) + half;   // k = tap*16 + c
                    const int c0  = hs * 8;
                    v8h t;
                    bool ok = false;
                    if (tap < 9) {
                        const int yy = y + tap / 3 - 1;
                        const int xx = x + tap % 3 - 1;
                        if (yy >= 0 && yy < 28 && xx >= 0 && xx < 28) {
                            t = *(const v8h*)(gh + ((yy * 28 + xx) << 4) + c0);
                            ok = true;
                        }
                    }
                    if (!ok) {
                        #pragma unroll
                        for (int i = 0; i < 8; ++i) t[i] = (_Float16)0.f;
                        // k == 144 is the bias column: A value 1.0
                        if (tap == 9 && hs == 0) t[0] = (_Float16)1.f;
                    }
                    #pragma unroll
                    for (int i = 0; i < 8; ++i) a[half * 8 + i] = t[i];
                }
                aF[kc] = a;
            }

            // ---- conv GEMM (K=160, bias in K) + ReLU -> staging h1[16][128] ----
            #pragma unroll
            for (int nf = 0; nf < 8; ++nf) {
                const int f = nf * 16 + lr;           // B-fragment N column
                v8f c;
                #pragma unroll
                for (int r = 0; r < 8; ++r) c[r] = 0.f;
                #pragma unroll
                for (int kc = 0; kc < 5; ++kc) {
                    const v8h* bp = (const v8h*)(wT + f * KPAD + kc * 32 + hs * 8);
                    v8h blo = bp[0], bhi = bp[2];
                    v16h bf;
                    #pragma unroll
                    for (int i = 0; i < 8; ++i) { bf[i] = blo[i]; bf[8 + i] = bhi[i]; }
                    c = __builtin_amdgcn_wmma_f32_16x16x32_f16(
                            false, aF[kc], false, bf, (short)0, c, false, false);
                }
                #pragma unroll
                for (int r = 0; r < 8; ++r)
                    myStg[(hs * 8 + r) * F_CH + f] = (_Float16)fmaxf(c[r], 0.f);
            }

            // ---- load h1 A-fragments (4 K-chunks) ----
            v16h hF[4];
            #pragma unroll
            for (int kc = 0; kc < 4; ++kc) {
                const int k0 = kc * 32;
                const _Float16* row = myStg + lr * F_CH;
                v8h lo = *(const v8h*)(row + k0 + hs * 8);
                v8h hi = *(const v8h*)(row + k0 + 16 + hs * 8);
                v16h a;
                #pragma unroll
                for (int i = 0; i < 8; ++i) { a[i] = lo[i]; a[8 + i] = hi[i]; }
                hF[kc] = a;
            }

            // ---- MLP1 GEMM (128x128) + bias + ReLU -> staging h2[16][128] ----
            #pragma unroll
            for (int nd = 0; nd < 8; ++nd) {
                const int d = nd * 16 + lr;
                const float bias = b1s[d];
                v8f c;
                #pragma unroll
                for (int r = 0; r < 8; ++r) c[r] = bias;
                #pragma unroll
                for (int kc = 0; kc < 4; ++kc) {
                    const v8h* bp = (const v8h*)(w1t + d * F_CH + kc * 32 + hs * 8);
                    v8h blo = bp[0], bhi = bp[2];
                    v16h bf;
                    #pragma unroll
                    for (int i = 0; i < 8; ++i) { bf[i] = blo[i]; bf[8 + i] = bhi[i]; }
                    c = __builtin_amdgcn_wmma_f32_16x16x32_f16(
                            false, hF[kc], false, bf, (short)0, c, false, false);
                }
                #pragma unroll
                for (int r = 0; r < 8; ++r)
                    myStg[(hs * 8 + r) * F_CH + d] = (_Float16)fmaxf(c[r], 0.f);
            }

            // ---- load h2 A-fragments ----
            v16h h2F[4];
            #pragma unroll
            for (int kc = 0; kc < 4; ++kc) {
                const int k0 = kc * 32;
                const _Float16* row = myStg + lr * F_CH;
                v8h lo = *(const v8h*)(row + k0 + hs * 8);
                v8h hi = *(const v8h*)(row + k0 + 16 + hs * 8);
                v16h a;
                #pragma unroll
                for (int i = 0; i < 8; ++i) { a[i] = lo[i]; a[8 + i] = hi[i]; }
                h2F[kc] = a;
            }

            // ---- MLP2 GEMM (128x16) + bias -> delta[p][c] (f32) ----
            {
                const float bias = b2s[lr];
                v8f c;
                #pragma unroll
                for (int r = 0; r < 8; ++r) c[r] = bias;
                #pragma unroll
                for (int kc = 0; kc < 4; ++kc) {
                    const v8h* bp = (const v8h*)(w2t + lr * D_CH + kc * 32 + hs * 8);
                    v8h blo = bp[0], bhi = bp[2];
                    v16h bf;
                    #pragma unroll
                    for (int i = 0; i < 8; ++i) { bf[i] = blo[i]; bf[8 + i] = bhi[i]; }
                    c = __builtin_amdgcn_wmma_f32_16x16x32_f16(
                            false, h2F[kc], false, bf, (short)0, c, false, false);
                }
                #pragma unroll
                for (int r = 0; r < 8; ++r)
                    dlt[(tile * 16 + hs * 8 + r) * C_CH + lr] = c[r];
            }
        } // tiles

        __syncthreads();
        // ---- masked residual apply (ch0 invariant => alive read is safe) ----
        for (int i = tid; i < HWPIX * C_CH; i += 256) {
            int c = i & 15;
            if (c != 0) {
                int p = i >> 4;
                float alive = (g[p * C_CH] > 0.1f) ? 1.f : 0.f;
                float nv = g[i] + dlt[i] * alive;
                g[i]  = nv;
                gh[i] = (_Float16)nv;
            }
        }
        __syncthreads();
    } // steps

    // ---- softmax over channels 1..10 -> out ----
    for (int p = tid; p < HWPIX; p += 256) {
        const float* gp = g + p * C_CH;
        float m = gp[1];
        #pragma unroll
        for (int j = 2; j <= 10; ++j) m = fmaxf(m, gp[j]);
        float e[10], s = 0.f;
        #pragma unroll
        for (int j = 0; j < 10; ++j) { e[j] = __expf(gp[1 + j] - m); s += e[j]; }
        float inv = 1.f / s;
        float* op = out + (b * HWPIX + p) * 10;
        #pragma unroll
        for (int j = 0; j < 10; ++j) op[j] = e[j] * inv;
    }
}

extern "C" void kernel_launch(void* const* d_in, const int* in_sizes, int n_in,
                              void* d_out, int out_size, void* d_ws, size_t ws_size,
                              hipStream_t stream) {
    const float* in     = (const float*)d_in[0];
    const float* conv_w = (const float*)d_in[1];
    const float* conv_b = (const float*)d_in[2];
    const float* w1     = (const float*)d_in[3];
    const float* b1     = (const float*)d_in[4];
    const float* w2     = (const float*)d_in[5];
    const float* b2     = (const float*)d_in[6];
    float* out = (float*)d_out;

    nca_fused_kernel<<<256, 256, SMEM_BYTES, stream>>>(
        in, conv_w, conv_b, w1, b1, w2, b2, out);
}
